// GPT2AttentionXWWX_56994216018543
// MI455X (gfx1250) — matrix operand order repaired and checked by reading, not compiled, not run on hardware
//
#include <hip/hip_runtime.h>

// GPT-2 attention (XWWX form == standard attention with K = x@W_k + b_k)
// B=2, S=2048, E=1024, H=16, DH=64. All GEMMs via v_wmma_f32_16x16x32_bf16.
// Tile staging uses GLOBAL_LOAD_ASYNC_TO_LDS (ASYNCcnt) where available.

#define E_DIM 1024
#define NHEAD 16
#define DHEAD 64
#define BATCH 2
#define SEQ   2048
#define NTOK  (BATCH * SEQ)   // 4096

typedef __attribute__((ext_vector_type(16))) __bf16       v16bf;
typedef __attribute__((ext_vector_type(8)))  float        v8f;
typedef __attribute__((ext_vector_type(8)))  unsigned int v8u;
typedef __attribute__((ext_vector_type(4)))  int          v4i;

#if __has_builtin(__builtin_amdgcn_global_load_async_to_lds_b128) && \
    __has_builtin(__builtin_amdgcn_s_wait_asynccnt)
#define ASYNC_LDS 1
typedef __attribute__((address_space(1))) v4i* as1_v4i_p;
typedef __attribute__((address_space(3))) v4i* as3_v4i_p;
#endif

__device__ __forceinline__ unsigned short f2bf(float f) {
  unsigned int u = __float_as_uint(f);
  u += 0x7fffu + ((u >> 16) & 1u);     // round-to-nearest-even
  return (unsigned short)(u >> 16);
}

__device__ __forceinline__ v8f wmma_bf16(const v8u &a, const v8u &b, const v8f &c) {
  return __builtin_amdgcn_wmma_f32_16x16x32_bf16(
      false, __builtin_bit_cast(v16bf, a),
      false, __builtin_bit_cast(v16bf, b),
      (short)0, c, false, false);
}

// 16-byte global -> LDS copy; async (ASYNCcnt) when the toolchain exposes it.
__device__ __forceinline__ void cp16_g2l(const unsigned short* g, unsigned short* l) {
#ifdef ASYNC_LDS
  __builtin_amdgcn_global_load_async_to_lds_b128(
      (as1_v4i_p)(uintptr_t)g, (as3_v4i_p)(uintptr_t)l, 0, 0);
#else
  *(uint4*)l = *(const uint4*)g;
#endif
}
__device__ __forceinline__ void cp_wait() {
#ifdef ASYNC_LDS
  __builtin_amdgcn_s_wait_asynccnt(0);
#endif
}

// ---------------------------------------------------------------- convert
__global__ void cvt_f32_bf16(const float* __restrict__ src,
                             unsigned short* __restrict__ dst, int n) {
  int i = blockIdx.x * blockDim.x + threadIdx.x;
  int stride = gridDim.x * blockDim.x;
  for (; i < n; i += stride) dst[i] = f2bf(src[i]);
}

// ---------------------------------------------------------------- GEMM
// C[M=4096, N] = A[M,1024] @ Bw[1024,N] + bias.
// MODE 0: scatter bf16 into head-major Q/K/V.  MODE 1: fp32 store to out.
// Block tile 128x128, 8 waves (2x4 WMMA tiles each), K-chunks of 32 via LDS.
template <int MODE>
__global__ __launch_bounds__(256) void gemm_wmma(
    const unsigned short* __restrict__ A,
    const unsigned short* __restrict__ Bw, int ldb,
    const float* __restrict__ bias,
    unsigned short* __restrict__ qout, unsigned short* __restrict__ kout,
    unsigned short* __restrict__ vout, float* __restrict__ out) {
  __shared__ unsigned short At[128 * 32];  // [row][k]
  __shared__ unsigned short Bt[128 * 32];  // transposed: [n][k]

  const int tid  = threadIdx.x;
  const int lane = tid & 31;
  const int wave = tid >> 5;
  const int half = lane >> 4;
  const int ln   = lane & 15;
  const int wr = wave >> 1, wc = wave & 1;   // 4 x 2 wave grid
  const int row0 = blockIdx.y * 128;
  const int col0 = blockIdx.x * 128;

  v8f acc[2][4] = {};

  for (int k0 = 0; k0 < E_DIM; k0 += 32) {
    __syncthreads();
    {  // stage A: 128x32 bf16, two async 16B copies per thread
      int r = tid >> 1, hh = tid & 1;
      const unsigned short* src = A + (size_t)(row0 + r) * E_DIM + k0 + hh * 16;
      unsigned short* dst = At + r * 32 + hh * 16;
      cp16_g2l(src, dst);
      cp16_g2l(src + 8, dst + 8);
      if (k0 + 32 < E_DIM)  // speculative prefetch of next K tile
        __builtin_prefetch(src + 32, 0, 3);
    }
    {  // stage B transposed: Bw[k][n] -> Bt[n][k]  (manual: transpose on store)
#pragma unroll
      for (int i = 0; i < 2; ++i) {
        int kk = (tid >> 4) + i * 16, ng = (tid & 15) * 8;
        uint4 d = *(const uint4*)(Bw + (size_t)(k0 + kk) * ldb + col0 + ng);
        const unsigned short* ds = (const unsigned short*)&d;
#pragma unroll
        for (int j = 0; j < 8; ++j) Bt[(ng + j) * 32 + kk] = ds[j];
      }
    }
    cp_wait();
    __syncthreads();

    v8u af[2], bfr[4];
#pragma unroll
    for (int mi = 0; mi < 2; ++mi) {  // A frag: lane m=ln, K pairs per ISA layout
      const unsigned short* arow = At + (wr * 32 + mi * 16 + ln) * 32;
#pragma unroll
      for (int v = 0; v < 8; ++v) {
        int g = v >> 2, j = v & 3;
        af[mi][v] = *(const unsigned int*)(arow + g * 16 + half * 8 + j * 2);
      }
    }
#pragma unroll
    for (int ni = 0; ni < 4; ++ni) {  // B frag: lane n=ln, K pairs per ISA layout
      const unsigned short* brow = Bt + (wc * 64 + ni * 16 + ln) * 32;
#pragma unroll
      for (int v = 0; v < 8; ++v)
        bfr[ni][v] = *(const unsigned int*)(brow + (v >> 2) * 16 + half * 8 + (v & 3) * 2);
    }
#pragma unroll
    for (int mi = 0; mi < 2; ++mi)
#pragma unroll
      for (int ni = 0; ni < 4; ++ni)
        acc[mi][ni] = wmma_bf16(af[mi], bfr[ni], acc[mi][ni]);
  }

  // Epilogue: C layout -> row M = r + 8*half, col N = ln
#pragma unroll
  for (int mi = 0; mi < 2; ++mi) {
#pragma unroll
    for (int ni = 0; ni < 4; ++ni) {
      const int c = col0 + wc * 64 + ni * 16 + ln;
      const float bv = bias[c];
#pragma unroll
      for (int r = 0; r < 8; ++r) {
        const int t = row0 + wr * 32 + mi * 16 + r + 8 * half;
        const float val = acc[mi][ni][r] + bv;
        if (MODE == 0) {
          const int sec = c >> 10;           // 0=Q 1=K 2=V
          const int e = c & 1023;
          const int h = e >> 6, d = e & 63;
          const int b = t >> 11, s = t & 2047;
          const size_t idx = (((size_t)(b * NHEAD + h)) * SEQ + s) * DHEAD + d;
          unsigned short* dstp = (sec == 0) ? qout : (sec == 1 ? kout : vout);
          dstp[idx] = f2bf(val);
        } else {
          out[(size_t)t * E_DIM + c] = val;
        }
      }
    }
  }
}

// ---------------------------------------------------------------- attention
// One block per (128 query rows, head, batch); 8 waves x 16 query rows each.
// Flash-style online softmax; K/V tiles of 32 staged in LDS cooperatively.
__global__ __launch_bounds__(256) void attn_kernel(
    const unsigned short* __restrict__ qb, const unsigned short* __restrict__ kb,
    const unsigned short* __restrict__ vb, unsigned short* __restrict__ attnb) {
  __shared__ unsigned short Kt[32 * 64];      // [key][dh]
  __shared__ unsigned short Vt[64 * 32];      // transposed [dh][key]
  __shared__ unsigned short Pb[8 * 16 * 32];  // per-wave P scratch [q][key]

  const int tid  = threadIdx.x;
  const int lane = tid & 31, wave = tid >> 5;
  const int half = lane >> 4, ln = lane & 15;
  const int row0  = blockIdx.x * 128;
  const int hh    = blockIdx.y, bbb = blockIdx.z;
  const size_t headbase = ((size_t)(bbb * NHEAD + hh)) * SEQ * DHEAD;
  const int wrow0 = row0 + wave * 16;

  // Q fragments (16 rows x 64 dh => two 16x32 A-fragments), gathered per lane
  v8u qf[2];
  {
    const unsigned short* qrow = qb + headbase + (size_t)(wrow0 + ln) * DHEAD;
#pragma unroll
    for (int c = 0; c < 2; ++c)
#pragma unroll
      for (int v = 0; v < 8; ++v) {
        int g = v >> 2, j = v & 3;
        qf[c][v] = *(const unsigned int*)(qrow + c * 32 + g * 16 + half * 8 + j * 2);
      }
  }

  v8f o[4] = {};
  float mstat[8], lstat[8];
#pragma unroll
  for (int r = 0; r < 8; ++r) { mstat[r] = -3.0e38f; lstat[r] = 0.0f; }

  const float SCL = 0.18033688011112042f;  // (1/sqrt(64)) * log2(e)
  const int kend = row0 + 128;

  for (int kt = 0; kt < kend; kt += 32) {
    __syncthreads();
    {  // cooperative stage of K (async, row-major) and V (manual, transposed)
      int key = tid >> 3, dg = (tid & 7) * 8;
      cp16_g2l(kb + headbase + (size_t)(kt + key) * DHEAD + dg, Kt + key * 64 + dg);
      const uint4 vd = *(const uint4*)(vb + headbase + (size_t)(kt + key) * DHEAD + dg);
      const unsigned short* vs = (const unsigned short*)&vd;
#pragma unroll
      for (int i = 0; i < 8; ++i) Vt[(dg + i) * 32 + key] = vs[i];
    }
    cp_wait();
    __syncthreads();

    if (kt <= wrow0 + 15) {  // wave-uniform causal skip (EXEC stays all-1s)
      // S = Q @ K^T : two 16x16 score tiles, contraction over DH=64
      v8f s[2] = {};
#pragma unroll
      for (int tk = 0; tk < 2; ++tk) {
#pragma unroll
        for (int c = 0; c < 2; ++c) {
          v8u kfr;
          const unsigned short* krow = Kt + (tk * 16 + ln) * 64 + c * 32;
#pragma unroll
          for (int v = 0; v < 8; ++v)
            kfr[v] = *(const unsigned int*)(krow + (v >> 2) * 16 + half * 8 + (v & 3) * 2);
          s[tk] = wmma_bf16(qf[c], kfr, s[tk]);
        }
      }
      // causal mask in C-layout: row = wrow0 + r + 8*half, col key = kt + tk*16 + ln
#pragma unroll
      for (int tk = 0; tk < 2; ++tk)
#pragma unroll
        for (int r = 0; r < 8; ++r)
          if (kt + tk * 16 + ln > wrow0 + r + 8 * half) s[tk][r] = -3.0e38f;

      // online softmax update (row reductions across the 16-lane column groups)
      float p0[8], p1[8];
#pragma unroll
      for (int r = 0; r < 8; ++r) {
        float mx = fmaxf(s[0][r], s[1][r]);
#pragma unroll
        for (int off = 1; off < 16; off <<= 1)
          mx = fmaxf(mx, __shfl_xor(mx, off, 32));
        const float mnew = fmaxf(mstat[r], mx);
        const float corr = exp2f((mstat[r] - mnew) * SCL);
        const float a0 = exp2f((s[0][r] - mnew) * SCL);
        const float a1 = exp2f((s[1][r] - mnew) * SCL);
        float rs = a0 + a1;
#pragma unroll
        for (int off = 1; off < 16; off <<= 1)
          rs += __shfl_xor(rs, off, 32);
        lstat[r] = lstat[r] * corr + rs;
        mstat[r] = mnew;
        p0[r] = a0;
        p1[r] = a1;
#pragma unroll
        for (int tn = 0; tn < 4; ++tn) o[tn][r] *= corr;
      }

      // redistribute P: C-layout -> A-layout via per-wave LDS
      unsigned short* pw = Pb + wave * 16 * 32;
#pragma unroll
      for (int r = 0; r < 8; ++r) {
        pw[(r + 8 * half) * 32 + ln]      = f2bf(p0[r]);
        pw[(r + 8 * half) * 32 + 16 + ln] = f2bf(p1[r]);
      }
      asm volatile("s_wait_dscnt 0" ::: "memory");  // same-wave LDS RAW fence
      v8u pf;
      const unsigned short* prow = pw + ln * 32;
#pragma unroll
      for (int v = 0; v < 8; ++v) {
        int g = v >> 2, j = v & 3;
        pf[v] = *(const unsigned int*)(prow + g * 16 + half * 8 + j * 2);
      }

      // O += P @ V  (contraction over the 32 keys, 4 output col tiles)
#pragma unroll
      for (int tn = 0; tn < 4; ++tn) {
        v8u vfr;
        const unsigned short* vrow = Vt + (tn * 16 + ln) * 32;
#pragma unroll
        for (int v = 0; v < 8; ++v)
          vfr[v] = *(const unsigned int*)(vrow + (v >> 2) * 16 + half * 8 + (v & 3) * 2);
        o[tn] = wmma_bf16(pf, vfr, o[tn]);
      }
    }
  }

  // epilogue: O / l -> bf16 attn buffer in [B][S][E] layout
#pragma unroll
  for (int r = 0; r < 8; ++r) {
    const float inv = 1.0f / lstat[r];
    const int qrow = wrow0 + r + 8 * half;
    const int b = qrow >> 11, s_ = qrow & 2047;
    const size_t base = ((size_t)(b * SEQ + s_)) * E_DIM + hh * DHEAD;
#pragma unroll
    for (int tn = 0; tn < 4; ++tn)
      attnb[base + tn * 16 + ln] = f2bf(o[tn][r] * inv);
  }
}

// ---------------------------------------------------------------- launch
extern "C" void kernel_launch(void* const* d_in, const int* in_sizes, int n_in,
                              void* d_out, int out_size, void* d_ws, size_t ws_size,
                              hipStream_t stream) {
  const float* x     = (const float*)d_in[0];  // [B,S,E]
  const float* attw  = (const float*)d_in[1];  // [E,3E]
  const float* attb  = (const float*)d_in[2];  // [3E]
  const float* projw = (const float*)d_in[3];  // [E,E]
  const float* projb = (const float*)d_in[4];  // [E]
  float* out = (float*)d_out;

  unsigned short* ws   = (unsigned short*)d_ws;
  unsigned short* xb   = ws;                                  // 4096*1024
  unsigned short* wab  = xb + (size_t)NTOK * E_DIM;           // 1024*3072
  unsigned short* wpb  = wab + (size_t)E_DIM * 3 * E_DIM;     // 1024*1024
  unsigned short* qbuf = wpb + (size_t)E_DIM * E_DIM;         // head-major
  unsigned short* kbuf = qbuf + (size_t)NTOK * E_DIM;
  unsigned short* vbuf = kbuf + (size_t)NTOK * E_DIM;
  unsigned short* abuf = vbuf + (size_t)NTOK * E_DIM;         // [B,S,E] bf16

  cvt_f32_bf16<<<1024, 256, 0, stream>>>(x, xb, NTOK * E_DIM);
  cvt_f32_bf16<<<1024, 256, 0, stream>>>(attw, wab, E_DIM * 3 * E_DIM);
  cvt_f32_bf16<<<512, 256, 0, stream>>>(projw, wpb, E_DIM * E_DIM);

  gemm_wmma<0><<<dim3(3 * E_DIM / 128, NTOK / 128), 256, 0, stream>>>(
      xb, wab, 3 * E_DIM, attb, qbuf, kbuf, vbuf, nullptr);

  attn_kernel<<<dim3(SEQ / 128, NHEAD, BATCH), 256, 0, stream>>>(qbuf, kbuf, vbuf, abuf);

  gemm_wmma<1><<<dim3(E_DIM / 128, NTOK / 128), 256, 0, stream>>>(
      abuf, wpb, E_DIM, projb, nullptr, nullptr, nullptr, out);
}
